// _CNNCrossAttention_12000138625645
// MI455X (gfx1250) — matrix operand order repaired and checked by reading, not compiled, not run on hardware
//
#include <hip/hip_runtime.h>
#include <hip/hip_bf16.h>

// ---------------------------------------------------------------------------
// CNN cross-attention for MI455X (gfx1250), bf16 WMMA pipeline.
//   B=8, C_IN=C_OUT=256, H=W=48, N=2304
//   proj:  Q = Wq x1 + bq ; K = Wk x2 + bk ; V = Wv x2 + bv   (bf16 out)
//   attn:  O = softmax(Q K^T) V                               (flash-style)
// ---------------------------------------------------------------------------

constexpr int BB  = 8;
constexpr int CIN = 256;
constexpr int CC  = 256;   // C_OUT
constexpr int HW  = 48 * 48; // N = 2304

typedef __attribute__((ext_vector_type(16))) __bf16 v16bf;
typedef __attribute__((ext_vector_type(8)))  float  v8f;
typedef __attribute__((ext_vector_type(4)))  int    v4i;

#if __has_builtin(__builtin_amdgcn_global_load_async_to_lds_b128)
#define USE_ASYNC_COPY 1
#else
#define USE_ASYNC_COPY 0
#endif

__device__ __forceinline__ unsigned short f2bf(float f) {
  unsigned int u = __builtin_bit_cast(unsigned int, f);
  u += 0x7fffu + ((u >> 16) & 1u);            // round-to-nearest-even
  return (unsigned short)(u >> 16);
}

__device__ __forceinline__ v16bf make_frag(uint4 lo, uint4 hi) {
  struct P { uint4 lo, hi; } p{lo, hi};
  return __builtin_bit_cast(v16bf, p);
}

__device__ __forceinline__ v8f wmma_bf16(v16bf a, v16bf b, v8f c) {
  // D(f32 16x16) = A(bf16 16x32) * B(bf16 32x16) + C
  return __builtin_amdgcn_wmma_f32_16x16x32_bf16(
      /*neg_a=*/false, a, /*neg_b=*/false, b,
      /*c_mod=*/(short)0, c, /*reuse_a=*/false, /*reuse_b=*/false);
}

#if USE_ASYNC_COPY
// async 16B copy global -> LDS (tracked by ASYNCcnt)
__device__ __forceinline__ void async_copy16(const unsigned short* g, unsigned short* l) {
  __builtin_amdgcn_global_load_async_to_lds_b128(
      (__attribute__((address_space(1))) v4i*)(void*)g,
      (__attribute__((address_space(3))) v4i*)(void*)l, 0, 0);
}
#endif

// ---------------------------------------------------------------------------
// Projection: out = X^T W^T + bias, bf16 output, layout chosen at compile time.
//   TRANSPOSED=0: out[b][n][c]   (for Q, K)
//   TRANSPOSED=1: out[b][c][n]   (for V -> attention stages it as pure copies)
// Block: 128 threads (4 waves). Output tile: 64 n x 64 c. K loop: 8 x 32.
// ---------------------------------------------------------------------------
template <int TRANSPOSED>
__global__ __launch_bounds__(128) void proj_kernel(
    const float* __restrict__ X,     // [B][CIN][HW]
    const float* __restrict__ W,     // [CC][CIN]
    const float* __restrict__ bias,  // [CC]
    unsigned short* __restrict__ out) {
  __shared__ unsigned short XA[64 * 40]; // A tile: [n][k], stride 40 (80B, 16B-aligned)
  __shared__ unsigned short WT[64 * 32]; // Bt tile: [c][k], stride 32 (64B)

  const int nt = blockIdx.x, ct = blockIdx.y, b = blockIdx.z;
  const int n0 = nt * 64, c0 = ct * 64;
  const int t = threadIdx.x;
  const int lane = t & 31, wave = t >> 5;
  const int half = lane >> 4, l16 = lane & 15;

  const float* Xb = X + (size_t)b * CIN * HW;

  v8f acc[4] = {};  // 4 channel tiles (16 cols each), rows = this wave's 16 n

  for (int k0 = 0; k0 < CIN; k0 += 32) {
    __syncthreads();
    // stage X^T tile [64 n][32 k] as bf16 (coalesced along n)
    #pragma unroll
    for (int i = 0; i < 16; ++i) {
      int e = i * 128 + t;            // 0..2047
      int kk = e >> 6, n = e & 63;
      float f = Xb[(size_t)(k0 + kk) * HW + n0 + n];
      XA[n * 40 + kk] = f2bf(f);
    }
    // stage W tile [64 c][32 k] as bf16 (contiguous k, packed pairs)
    #pragma unroll
    for (int i = 0; i < 8; ++i) {
      int e = i * 128 + t;            // 0..1023 pairs
      int c = e >> 4, kk = (e & 15) * 2;
      const float* wp = W + (size_t)(c0 + c) * CIN + k0 + kk;
      unsigned int p = (unsigned int)f2bf(wp[0]) | ((unsigned int)f2bf(wp[1]) << 16);
      *(unsigned int*)&WT[c * 32 + kk] = p;
    }
    __syncthreads();

    // A fragment: lane holds row m = wave*16 + l16; K = half*16 .. +15 contiguous, as in B
    const unsigned short* ar = &XA[(wave * 16 + l16) * 40];
    v16bf af = make_frag(*(const uint4*)(ar + half * 16),
                         *(const uint4*)(ar + half * 16 + 8));
    #pragma unroll
    for (int cti = 0; cti < 4; ++cti) {
      // B fragment: lane holds column c; K = half*16 .. +15 contiguous
      const unsigned short* br = &WT[(cti * 16 + l16) * 32 + half * 16];
      v16bf bfr = make_frag(*(const uint4*)br, *(const uint4*)(br + 8));
      acc[cti] = wmma_bf16(af, bfr, acc[cti]);
    }
  }

  // epilogue: add bias, store bf16
  #pragma unroll
  for (int cti = 0; cti < 4; ++cti) {
    int c = c0 + cti * 16 + l16;
    float bv = bias[c];
    #pragma unroll
    for (int r = 0; r < 8; ++r) {
      int n = n0 + wave * 16 + r + half * 8;   // C/D layout: VGPR r -> rows r / r+8
      float v = acc[cti][r] + bv;
      if (TRANSPOSED)
        out[(size_t)(b * CC + c) * HW + n] = f2bf(v);
      else
        out[(size_t)(b * HW + n) * CC + c] = f2bf(v);
    }
  }
}

// ---------------------------------------------------------------------------
// Flash-style attention. Block: 128 threads (4 waves), 64 query rows/block
// (16 per wave). Key blocks of 64. K fragments streamed from global (shared
// lines served by WGP$ across the 4 waves), V staged in LDS (pre-transposed
// [c][n] in global -> pure 16B copies, async-to-LDS when available).
// ---------------------------------------------------------------------------
constexpr int VSTR = 72;  // VL row stride (elems): 144B, 16B-aligned, bank step 36
                          // -> the 16 lanes of a B-frag ds_load_b128 hit 16 banks

__global__ __launch_bounds__(128) void attn_kernel(
    const unsigned short* __restrict__ Q,   // bf16 [B][HW][CC]
    const unsigned short* __restrict__ K,   // bf16 [B][HW][CC]
    const unsigned short* __restrict__ Vt,  // bf16 [B][CC][HW]
    float* __restrict__ out) {              // f32  [B][CC][HW]
  __shared__ unsigned short VL[256 * VSTR]; // V block, Bt layout: [c][key], ~36KB
  __shared__ unsigned short PL[4][16 * 64]; // per-wave P staging, 8KB

  const int qt = blockIdx.x, b = blockIdx.z;
  const int t = threadIdx.x;
  const int lane = t & 31, wave = t >> 5;
  const int half = lane >> 4, l16 = lane & 15;
  const int qrow0 = qt * 64 + wave * 16;

  const unsigned short* Qb = Q  + (size_t)b * HW * CC;
  const unsigned short* Kb = K  + (size_t)b * HW * CC;
  const unsigned short* Vb = Vt + (size_t)b * CC * HW;

  // Q fragments for this wave's 16 rows, all 8 K-chunks (K dim = 256 channels)
  v16bf qf[8];
  {
    const unsigned short* qr = Qb + (size_t)(qrow0 + l16) * CC;
    #pragma unroll
    for (int ck = 0; ck < 8; ++ck)
      qf[ck] = make_frag(*(const uint4*)(qr + ck * 32 + half * 16),
                         *(const uint4*)(qr + ck * 32 + half * 16 + 8));
  }

  v8f o[16] = {};        // O accumulator: 16 channel tiles x (16 rows x 16 cols)
  float m[8], l[8];
  #pragma unroll
  for (int r = 0; r < 8; ++r) { m[r] = -1e30f; l[r] = 0.0f; }

  for (int key0 = 0; key0 < HW; key0 += 64) {
    __syncthreads();
    // ---- stage V block into LDS: VL[c][key] (16B chunks; async when available)
    #pragma unroll
    for (int i = 0; i < 16; ++i) {
      int e = i * 128 + t;
      int c = e >> 3, part = e & 7;
      const unsigned short* g = Vb + (size_t)c * HW + key0 + part * 8;
      unsigned short* dst = &VL[c * VSTR + part * 8];
#if USE_ASYNC_COPY
      async_copy16(g, dst);
#else
      *(uint4*)dst = *(const uint4*)g;
#endif
    }
#if USE_ASYNC_COPY
#if __has_builtin(__builtin_amdgcn_s_wait_asynccnt)
    __builtin_amdgcn_s_wait_asynccnt(0);
#else
    asm volatile("s_wait_asynccnt 0" ::: "memory");
#endif
#endif
    __syncthreads();

    // ---- S = Q K^T for 4 key tiles of 16 (K frags direct from global/WGP$) ----
    v8f s[4];
    #pragma unroll
    for (int kt = 0; kt < 4; ++kt) {
      v8f sa = {};
      const unsigned short* kr =
          Kb + (size_t)(key0 + kt * 16 + l16) * CC + half * 16;
      #pragma unroll
      for (int ck = 0; ck < 8; ++ck) {
        v16bf kf = make_frag(*(const uint4*)(kr + ck * 32),
                             *(const uint4*)(kr + ck * 32 + 8));
        sa = wmma_bf16(qf[ck], kf, sa);
      }
      s[kt] = sa;
    }

    // ---- online softmax (per-row stats live in C/D layout: VGPR r -> row r/r+8)
    float alpha[8];
    #pragma unroll
    for (int r = 0; r < 8; ++r) {
      float v = fmaxf(fmaxf(s[0][r], s[1][r]), fmaxf(s[2][r], s[3][r]));
      v = fmaxf(v, __shfl_xor(v, 1));
      v = fmaxf(v, __shfl_xor(v, 2));
      v = fmaxf(v, __shfl_xor(v, 4));
      v = fmaxf(v, __shfl_xor(v, 8));   // row max within 16-lane half
      float mn = fmaxf(m[r], v);
      alpha[r] = __expf(m[r] - mn);
      m[r] = mn;
    }
    // exponentiate + write P (bf16) to per-wave LDS in row-major [row][key]
    unsigned short* pw = &PL[wave][0];
    #pragma unroll
    for (int kt = 0; kt < 4; ++kt) {
      #pragma unroll
      for (int r = 0; r < 8; ++r) {
        float p = __expf(s[kt][r] - m[r]);
        s[kt][r] = p;
        pw[(r + half * 8) * 64 + kt * 16 + l16] = f2bf(p);
      }
    }
    #pragma unroll
    for (int r = 0; r < 8; ++r) {
      float rs = s[0][r] + s[1][r] + s[2][r] + s[3][r];
      rs += __shfl_xor(rs, 1);
      rs += __shfl_xor(rs, 2);
      rs += __shfl_xor(rs, 4);
      rs += __shfl_xor(rs, 8);
      l[r] = l[r] * alpha[r] + rs;
    }
    // rescale O
    #pragma unroll
    for (int ct2 = 0; ct2 < 16; ++ct2)
      #pragma unroll
      for (int r = 0; r < 8; ++r)
        o[ct2][r] *= alpha[r];

    // ---- read P back as A-fragments (wave-private LDS, same-wave ordering) ----
    v16bf pf[2];
    {
      const unsigned short* pr = &PL[wave][l16 * 64];
      #pragma unroll
      for (int kc = 0; kc < 2; ++kc)
        pf[kc] = make_frag(*(const uint4*)(pr + kc * 32 + half * 16),
                           *(const uint4*)(pr + kc * 32 + half * 16 + 8));
    }
    // ---- O += P * V  (B-frags from VL: [c][key], K dim = 64 keys, 2 chunks) ----
    #pragma unroll
    for (int ct2 = 0; ct2 < 16; ++ct2) {
      const unsigned short* vr = &VL[(ct2 * 16 + l16) * VSTR + half * 16];
      #pragma unroll
      for (int kc = 0; kc < 2; ++kc) {
        v16bf vf = make_frag(*(const uint4*)(vr + kc * 32),
                             *(const uint4*)(vr + kc * 32 + 8));
        o[ct2] = wmma_bf16(pf[kc], vf, o[ct2]);
      }
    }
  }

  // ---- epilogue: normalize, store fp32 [B][C][N] ----
  float inv[8];
  #pragma unroll
  for (int r = 0; r < 8; ++r) inv[r] = 1.0f / l[r];
  float* ob = out + (size_t)b * CC * HW;
  #pragma unroll
  for (int ct2 = 0; ct2 < 16; ++ct2) {
    int c = ct2 * 16 + l16;
    #pragma unroll
    for (int r = 0; r < 8; ++r) {
      int n = qrow0 + r + half * 8;
      ob[(size_t)c * HW + n] = o[ct2][r] * inv[r];
    }
  }
}

// ---------------------------------------------------------------------------
extern "C" void kernel_launch(void* const* d_in, const int* in_sizes, int n_in,
                              void* d_out, int out_size, void* d_ws, size_t ws_size,
                              hipStream_t stream) {
  const float* x1 = (const float*)d_in[0];
  const float* x2 = (const float*)d_in[1];
  const float* Wq = (const float*)d_in[2];
  const float* bq = (const float*)d_in[3];
  const float* Wk = (const float*)d_in[4];
  const float* bk = (const float*)d_in[5];
  const float* Wv = (const float*)d_in[6];
  const float* bv = (const float*)d_in[7];
  float* out = (float*)d_out;

  const size_t per = (size_t)BB * HW * CC;   // elements per bf16 tensor
  unsigned short* qw = (unsigned short*)d_ws;
  unsigned short* kw = qw + per;
  unsigned short* vw = kw + per;             // 3*per*2 = ~28.3 MB of workspace

  dim3 pg(HW / 64, CC / 64, BB), pb(128);
  proj_kernel<0><<<pg, pb, 0, stream>>>(x1, Wq, bq, qw);  // Q: [b][n][c]
  proj_kernel<0><<<pg, pb, 0, stream>>>(x2, Wk, bk, kw);  // K: [b][n][c]
  proj_kernel<1><<<pg, pb, 0, stream>>>(x2, Wv, bv, vw);  // V: [b][c][n] (pre-transposed)

  dim3 ag(HW / 64, 1, BB), ab(128);
  attn_kernel<<<ag, ab, 0, stream>>>(qw, kw, vw, out);
}
